// BahdanauDecoder_4647154615035
// MI455X (gfx1250) — matrix operand order repaired and checked by reading, not compiled, not run on hardware
//
#include <hip/hip_runtime.h>

// ---------------------------------------------------------------------------
// Bahdanau attention GRU decoder for MI455X (gfx1250, wave32, WMMA).
// - src_enc & enc_proj kept as f16 (67MB each) -> both L2-resident (192MB L2),
//   so the 127 sequential steps stream attention tensors from L2, not HBM.
// - All GEMMs: v_wmma_f32_16x16x32_f16, weights pre-transposed to (K,N) f16.
// - N is a template constant -> strength-reduced B addressing (no v_mul_u64).
// - All 6 loads of a k-chunk issued before the 4 WMMAs -> partial loadcnt waits.
// - Wo/Wz/Wr fused into one (K=1280, N=1536) GEMM whose epilogue writes ti to
//   d_out, z to ws, and r*s directly into xn (f16).
// ---------------------------------------------------------------------------

typedef _Float16 h16;
typedef __attribute__((ext_vector_type(16))) _Float16 v16h;
typedef __attribute__((ext_vector_type(8)))  _Float16 v8h;
typedef __attribute__((ext_vector_type(8)))  float    v8f;

#define SRC_LEN 256
#define TGT_LEN 128
#define BATCH   256
#define NH      512
#define NE      256
#define NX      1280   // NE + 2*NH
#define NSTEP   127

// ---------------- small helpers -------------------------------------------

__device__ __forceinline__ float warp_sum32(float v) {
#pragma unroll
  for (int m = 16; m >= 1; m >>= 1) v += __shfl_xor(v, m, 32);
  return v;
}
__device__ __forceinline__ float warp_max32(float v) {
#pragma unroll
  for (int m = 16; m >= 1; m >>= 1) v = fmaxf(v, __shfl_xor(v, m, 32));
  return v;
}
__device__ __forceinline__ float fast_tanh(float x) {
  x = fminf(fmaxf(x, -15.f), 15.f);
  float e = __expf(2.f * x);
  return (e - 1.f) / (e + 1.f);
}
__device__ __forceinline__ float fast_sigmoid(float x) {
  x = fminf(fmaxf(x, -30.f), 30.f);
  return 1.f / (1.f + __expf(-x));
}

// ---------------- setup kernels -------------------------------------------

__global__ void k_cvt_f16(const float* __restrict__ in, h16* __restrict__ out, size_t n) {
  size_t i = (size_t)blockIdx.x * blockDim.x + threadIdx.x;
  if (i < n) out[i] = (h16)in[i];
}

// Bt[k*ldo + colo + n] = W[n*ldin + off + k]   (one-time, uncoalesced is fine)
__global__ void k_wtrans_f16(const float* __restrict__ W, h16* __restrict__ Bt,
                             int N, int K, int ldin, int off, int ldo, int colo) {
  size_t i = (size_t)blockIdx.x * blockDim.x + threadIdx.x;
  if (i >= (size_t)N * K) return;
  int k = (int)(i / N), n = (int)(i % N);
  Bt[(size_t)k * ldo + colo + n] = (h16)W[(size_t)n * ldin + off + k];
}

// emb_h[t*BATCH*NE + b*NE + e] = table[tgt[t*BATCH+b]*NE + e], t in [0,127)
__global__ void k_embed(const int* __restrict__ tgt, const float* __restrict__ table,
                        h16* __restrict__ emb_h) {
  int rb  = blockIdx.x;                 // t*BATCH + b
  int idx = tgt[rb];
  emb_h[(size_t)rb * NE + threadIdx.x] = (h16)table[(size_t)idx * NE + threadIdx.x];
}

// ---------------- WMMA main loop ------------------------------------------
// acc[4] (16 rows x 64 cols) over K; A row-major lda=K (f16), Bt (K,N) f16.
// ISA f16 A-frag 16x32: lane m<16 -> K {0..7,16..23}; lane m+16 -> {8..15,24..31}
// ISA f16 B-frag 32x16: lane n<16 -> col n, K {0..7,16..23}; lane n+16 -> {8..15,24..31}
template <int N>
__device__ __forceinline__ void wmma_mainloop(const h16* __restrict__ A,
                                              const h16* __restrict__ Bt,
                                              int row, int n0, int K,
                                              int lane, v8f acc[4]) {
  const int hsel = (lane >> 4) * 8;
  const h16* aptr = A + (size_t)row * K + hsel;
  const h16* bptr = Bt + (size_t)hsel * N + n0 + (lane & 15);
#pragma unroll 2
  for (int k0 = 0; k0 < K; k0 += 32) {
    v8h  alo = *(const v8h*)(aptr);
    v8h  ahi = *(const v8h*)(aptr + 16);
    v16h b0, b1, b2, b3;
#pragma unroll
    for (int i = 0; i < 16; ++i) {
      const h16* bp = bptr + (size_t)((i & 7) + ((i >> 3) << 4)) * N;
      b0[i] = bp[0]; b1[i] = bp[16]; b2[i] = bp[32]; b3[i] = bp[48];
    }
    aptr += 32;
    bptr += (size_t)32 * N;
    v16h af;
#pragma unroll
    for (int i = 0; i < 8; ++i) { af[i] = alo[i]; af[i + 8] = ahi[i]; }
    acc[0] = __builtin_amdgcn_wmma_f32_16x16x32_f16(false, af, false, b0, (short)0, acc[0], false, false);
    acc[1] = __builtin_amdgcn_wmma_f32_16x16x32_f16(false, af, false, b1, (short)0, acc[1], false, false);
    acc[2] = __builtin_amdgcn_wmma_f32_16x16x32_f16(false, af, false, b2, (short)0, acc[2], false, false);
    acc[3] = __builtin_amdgcn_wmma_f32_16x16x32_f16(false, af, false, b3, (short)0, acc[3], false, false);
  }
}

// Generic GEMM: C = act(A x Bt + bias). ACT: 0=none 1=tanh 3=GRU update.
// Block = 4 waves = 64 rows x 64 cols. grid = (N/64, M/64).
template <int N, int ACT>
__global__ __launch_bounds__(128)
void k_gemm_wmma(const h16* __restrict__ A, const h16* __restrict__ Bt,
                 const float* __restrict__ bias,
                 float* __restrict__ Cf, h16* __restrict__ Ch, int K,
                 const float* __restrict__ zbuf, const float* __restrict__ sold) {
  const int lane = threadIdx.x & 31;
  const int wave = threadIdx.x >> 5;
  const int m0   = blockIdx.y * 64 + wave * 16;
  const int n0   = blockIdx.x * 64;
  v8f acc[4] = {};
  wmma_mainloop<N>(A, Bt, m0 + (lane & 15), n0, K, lane, acc);

#pragma unroll
  for (int j = 0; j < 4; ++j) {
    int n = n0 + j * 16 + (lane & 15);
    float bb = bias ? bias[n] : 0.f;
#pragma unroll
    for (int r = 0; r < 8; ++r) {
      int m = m0 + r + ((lane >> 4) << 3);
      float v = acc[j][r] + bb;
      if (ACT == 1) v = fast_tanh(v);
      if (ACT == 3) {
        float nf = fast_tanh(v);
        float z  = zbuf[(size_t)m * N + n];
        float sv = sold[(size_t)m * N + n];
        v = (1.f - z) * sv + z * nf;
      }
      if (Cf) Cf[(size_t)m * N + n] = v;
      if (Ch) Ch[(size_t)m * N + n] = (h16)v;
    }
  }
}

// Fused Wo|Wz|Wr GEMM: A = x (256x1280), Bt = (1280 x 1536).
// Column segment 0: ti -> d_out; 1: z=sigmoid -> ws; 2: xn[...] = sigmoid(.)*s
__global__ __launch_bounds__(128)
void k_gemm_ozr(const h16* __restrict__ A, const h16* __restrict__ Bt,
                const float* __restrict__ bo, const float* __restrict__ bz,
                const float* __restrict__ br,
                float* __restrict__ ti_out, float* __restrict__ z_out,
                h16* __restrict__ xn_h, const float* __restrict__ s_f32) {
  constexpr int N = 3 * NH;
  const int lane = threadIdx.x & 31;
  const int wave = threadIdx.x >> 5;
  const int m0   = blockIdx.y * 64 + wave * 16;
  const int n0   = blockIdx.x * 64;
  v8f acc[4] = {};
  wmma_mainloop<N>(A, Bt, m0 + (lane & 15), n0, NX, lane, acc);

  const int which = n0 >> 9;                 // 64-col tile never crosses a 512 seg
  const float* bias = (which == 0) ? bo : (which == 1) ? bz : br;
#pragma unroll
  for (int j = 0; j < 4; ++j) {
    int n  = n0 + j * 16 + (lane & 15);
    int nl = n & (NH - 1);
    float bb = bias[nl];
#pragma unroll
    for (int r = 0; r < 8; ++r) {
      int m = m0 + r + ((lane >> 4) << 3);
      float v = acc[j][r] + bb;
      if (which == 0) {
        ti_out[(size_t)m * NH + nl] = v;
      } else if (which == 1) {
        z_out[(size_t)m * NH + nl] = fast_sigmoid(v);
      } else {
        float rr = fast_sigmoid(v);
        xn_h[(size_t)m * NX + NE + nl] = (h16)(rr * s_f32[(size_t)m * NH + nl]);
      }
    }
  }
}

// ---------------- per-step kernels ----------------------------------------

// dec_states[step]=s; x[:,0:NE]=emb; xn[:,0:NE]=emb; x[:,NE:NE+NH]=s
__global__ __launch_bounds__(512)
void k_step_prep(const float* __restrict__ s_f32, const h16* __restrict__ s_h,
                 const h16* __restrict__ emb_step, float* __restrict__ states_out,
                 h16* __restrict__ x_h, h16* __restrict__ xn_h) {
  int b = blockIdx.x, t = threadIdx.x;
  states_out[(size_t)b * NH + t] = s_f32[(size_t)b * NH + t];
  x_h[(size_t)b * NX + NE + t]   = s_h[(size_t)b * NH + t];
  if (t < NE) {
    h16 e = emb_step[(size_t)b * NE + t];
    x_h[(size_t)b * NX + t]  = e;
    xn_h[(size_t)b * NX + t] = e;
  }
}

// ei_t[b*SRC_LEN+s] = mask ? va . tanh(enc_proj[s,b,:]+sWa[b,:]) : -inf
__global__ __launch_bounds__(256)
void k_attn_scores(const h16* __restrict__ enc_proj, const float* __restrict__ sWa,
                   const float* __restrict__ va, const int* __restrict__ src_len,
                   float* __restrict__ ei_t) {
  int lane = threadIdx.x & 31, wave = threadIdx.x >> 5;
  int s = blockIdx.x;
  int b = blockIdx.y * 8 + wave;
  const h16*   ep = enc_proj + ((size_t)s * BATCH + b) * NH + lane * 16;
  const float* sw = sWa + (size_t)b * NH + lane * 16;
  v16h e = *(const v16h*)ep;
  float acc = 0.f;
#pragma unroll
  for (int i = 0; i < 16; ++i)
    acc += fast_tanh((float)e[i] + sw[i]) * va[lane * 16 + i];
  acc = warp_sum32(acc);
  if (lane == 0)
    ei_t[(size_t)b * SRC_LEN + s] = (s < src_len[b]) ? acc : -__builtin_inff();
}

// softmax over s per batch row; write attn weights to ws and to d_out
__global__ __launch_bounds__(32)
void k_attn_softmax(const float* __restrict__ ei_t, float* __restrict__ a_t,
                    float* __restrict__ attn_out) {
  int b = blockIdx.x, lane = threadIdx.x;
  const float* row = ei_t + (size_t)b * SRC_LEN;
  float v[8], mx = -__builtin_inff();
#pragma unroll
  for (int j = 0; j < 8; ++j) { v[j] = row[lane + 32 * j]; mx = fmaxf(mx, v[j]); }
  mx = warp_max32(mx);
  float sum = 0.f;
#pragma unroll
  for (int j = 0; j < 8; ++j) { v[j] = __expf(v[j] - mx); sum += v[j]; }
  sum = warp_sum32(sum);
  float inv = 1.f / sum;
#pragma unroll
  for (int j = 0; j < 8; ++j) {
    float a = v[j] * inv;
    a_t[(size_t)b * SRC_LEN + lane + 32 * j]      = a;
    attn_out[(size_t)b * SRC_LEN + lane + 32 * j] = a;
  }
}

// ci[b,h] = sum_s a[b,s] * src_enc[s,b,h]; write into x and xn tail slots
__global__ __launch_bounds__(256)
void k_attn_context(const h16* __restrict__ src_enc_h, const float* __restrict__ a_t,
                    h16* __restrict__ x_h, h16* __restrict__ xn_h) {
  int b = blockIdx.x, t = threadIdx.x;   // thread owns h = 2t, 2t+1
  const float* arow = a_t + (size_t)b * SRC_LEN;
  const h16* base = src_enc_h + (size_t)b * NH + 2 * t;
  float ax = 0.f, ay = 0.f;
  for (int s = 0; s < SRC_LEN; ++s) {
    float a = arow[s];                               // block-uniform -> SGPR
    const h16* p = base + (size_t)s * BATCH * NH;
    ax += a * (float)p[0];
    ay += a * (float)p[1];
  }
  size_t o = (size_t)b * NX + (NE + NH) + 2 * t;
  x_h[o]  = (h16)ax; x_h[o + 1]  = (h16)ay;
  xn_h[o] = (h16)ax; xn_h[o + 1] = (h16)ay;
}

// ---------------- host-side launcher --------------------------------------

extern "C" void kernel_launch(void* const* d_in, const int* in_sizes, int n_in,
                              void* d_out, int out_size, void* d_ws, size_t ws_size,
                              hipStream_t stream) {
  const float* src_enc = (const float*)d_in[0];
  const int*   tgt     = (const int*)d_in[1];
  const int*   src_len = (const int*)d_in[2];
  const float* emb_tbl = (const float*)d_in[3];
  const float* Ws_w = (const float*)d_in[4];  const float* Ws_b = (const float*)d_in[5];
  const float* Wz_w = (const float*)d_in[6];  const float* Wz_b = (const float*)d_in[7];
  const float* Wr_w = (const float*)d_in[8];  const float* Wr_b = (const float*)d_in[9];
  const float* Wn_w = (const float*)d_in[10]; const float* Wn_b = (const float*)d_in[11];
  const float* Wa_w = (const float*)d_in[12]; const float* Wa_b = (const float*)d_in[13];
  const float* va_w = (const float*)d_in[14];
  const float* Wo_w = (const float*)d_in[15]; const float* Wo_b = (const float*)d_in[16];

  float* out = (float*)d_out;
  float* out_outputs = out;                                     // (127,256,512)
  float* out_states  = out + (size_t)NSTEP * BATCH * NH;        // (127,256,512)
  float* out_attns   = out + (size_t)2 * NSTEP * BATCH * NH;    // (127,256,256)

  // ---- workspace carve-up ----
  char* w = (char*)d_ws;
  size_t off = 0;
  auto carve = [&](size_t bytes) { void* p = w + off; off += (bytes + 255) & ~(size_t)255; return p; };
  h16*   src_enc_h = (h16*)  carve((size_t)SRC_LEN * BATCH * NH * 2);   // 67 MB
  h16*   enc_proj  = (h16*)  carve((size_t)SRC_LEN * BATCH * NH * 2);   // 67 MB
  h16*   emb_h     = (h16*)  carve((size_t)NSTEP * BATCH * NE * 2);     // 16.6 MB
  h16*   WaS_T     = (h16*)  carve((size_t)NH * NH * 2);
  h16*   WaE_T     = (h16*)  carve((size_t)NH * NH * 2);
  h16*   Ws_T      = (h16*)  carve((size_t)NH * NH * 2);
  h16*   Wn_T      = (h16*)  carve((size_t)NX * NH * 2);
  h16*   Bt_ozr    = (h16*)  carve((size_t)NX * 3 * NH * 2);            // 3.9 MB
  float* s_f32     = (float*)carve((size_t)BATCH * NH * 4);
  h16*   s_h       = (h16*)  carve((size_t)BATCH * NH * 2);
  float* sWa       = (float*)carve((size_t)BATCH * NH * 4);
  float* ei_t      = (float*)carve((size_t)BATCH * SRC_LEN * 4);
  float* a_t       = (float*)carve((size_t)BATCH * SRC_LEN * 4);
  float* z_f32     = (float*)carve((size_t)BATCH * NH * 4);
  h16*   x_h       = (h16*)  carve((size_t)BATCH * NX * 2);
  h16*   xn_h      = (h16*)  carve((size_t)BATCH * NX * 2);
  (void)ws_size; (void)in_sizes; (void)n_in; (void)out_size;

  // ---- one-time setup ----
  {
    size_t n = (size_t)SRC_LEN * BATCH * NH;
    k_cvt_f16<<<dim3((unsigned)((n + 255) / 256)), 256, 0, stream>>>(src_enc, src_enc_h, n);
  }
  k_embed<<<NSTEP * BATCH, NE, 0, stream>>>(tgt, emb_tbl, emb_h);

  {
    int blk = 256;
    auto tr = [&](const float* W, h16* Bt, int N, int K, int ldin, int o, int ldo, int colo) {
      k_wtrans_f16<<<dim3((unsigned)(((size_t)N * K + blk - 1) / blk)), blk, 0, stream>>>(
          W, Bt, N, K, ldin, o, ldo, colo);
    };
    tr(Wa_w, WaS_T, NH, NH, 2 * NH, 0,  NH, 0);   // Wa_s = Wa_w[:, :NH]
    tr(Wa_w, WaE_T, NH, NH, 2 * NH, NH, NH, 0);   // Wa_e = Wa_w[:, NH:]
    tr(Ws_w, Ws_T,  NH, NH, NH, 0,      NH, 0);
    tr(Wn_w, Wn_T,  NH, NX, NX, 0,      NH, 0);
    tr(Wo_w, Bt_ozr, NH, NX, NX, 0, 3 * NH, 0);       // cols [0,512)
    tr(Wz_w, Bt_ozr, NH, NX, NX, 0, 3 * NH, NH);      // cols [512,1024)
    tr(Wr_w, Bt_ozr, NH, NX, NX, 0, 3 * NH, 2 * NH);  // cols [1024,1536)
  }

  // enc_proj = src_enc @ Wa_e.T + Wa_b   (M=65536, N=512, K=512) -> f16
  k_gemm_wmma<NH, 0><<<dim3(NH / 64, (SRC_LEN * BATCH) / 64), 128, 0, stream>>>(
      src_enc_h, WaE_T, Wa_b, nullptr, enc_proj, NH, nullptr, nullptr);

  // s0 = tanh(src_enc[0] @ Ws.T + Ws_b)  (M=256, N=512, K=512)
  k_gemm_wmma<NH, 1><<<dim3(NH / 64, BATCH / 64), 128, 0, stream>>>(
      src_enc_h, Ws_T, Ws_b, s_f32, s_h, NH, nullptr, nullptr);

  // ---- sequential decode loop ----
  dim3 g512(NH / 64, BATCH / 64);        // (8,4)
  dim3 gozr(3 * NH / 64, BATCH / 64);    // (24,4)
  for (int t = 0; t < NSTEP; ++t) {
    k_step_prep<<<BATCH, NH, 0, stream>>>(
        s_f32, s_h, emb_h + (size_t)t * BATCH * NE,
        out_states + (size_t)t * BATCH * NH, x_h, xn_h);

    // sWa = s @ Wa_s.T
    k_gemm_wmma<NH, 0><<<g512, 128, 0, stream>>>(
        s_h, WaS_T, nullptr, sWa, nullptr, NH, nullptr, nullptr);

    k_attn_scores<<<dim3(SRC_LEN, BATCH / 8), 256, 0, stream>>>(
        enc_proj, sWa, va_w, src_len, ei_t);

    k_attn_softmax<<<BATCH, 32, 0, stream>>>(
        ei_t, a_t, out_attns + (size_t)t * BATCH * SRC_LEN);

    k_attn_context<<<BATCH, NH / 2, 0, stream>>>(src_enc_h, a_t, x_h, xn_h);

    // fused: ti -> d_out, z -> ws, xn[:,NE:NE+NH] = sigmoid(r)*s
    k_gemm_ozr<<<gozr, 128, 0, stream>>>(
        x_h, Bt_ozr, Wo_b, Wz_b, Wr_b,
        out_outputs + (size_t)t * BATCH * NH, z_f32, xn_h, s_f32);

    // n = tanh(xn @ Wn.T + b);  s = (1-z)*s + z*n   (fused epilogue)
    k_gemm_wmma<NH, 3><<<g512, 128, 0, stream>>>(
        xn_h, Wn_T, Wn_b, s_f32, s_h, NX, z_f32, s_f32);
  }
}
